// AdaptivePooling_34308198760541
// MI455X (gfx1250) — matrix operand order repaired and checked by reading, not compiled, not run on hardware
//
#include <hip/hip_runtime.h>
#include <hip/hip_bf16.h>
#include <math.h>

// ---- types -----------------------------------------------------------------
typedef __bf16 bf16x16 __attribute__((ext_vector_type(16)));
typedef float  f32x8   __attribute__((ext_vector_type(8)));

#define H   256
#define HH  128      // H/2
#define C3H 768      // 3*H

// round-to-nearest-even f32 -> bf16 bits
__device__ __forceinline__ unsigned short f2bf(float f) {
    unsigned int u = __float_as_uint(f);
    unsigned int r = u + 0x7FFFu + ((u >> 16) & 1u);
    return (unsigned short)(r >> 16);
}

// ---- K0: pack W1 / Wf into bf16 WMMA-B fragment order ----------------------
// B (32x16, 16-bit), contiguous K order as packed here: lane L -> column n = L&15; half = L>>4;
// element e (VGPR e>>1, pos e&1) -> K = e + 16*half.
// fragW1 index: ((c*8 + kk)*32 + L)*16 + e       (c: 8 col tiles, kk: 8 K-steps)
// fragWf index: ((c*24 + kk)*32 + L)*16 + e      (c: 16 col tiles, kk: 24 K-steps)
__global__ void __launch_bounds__(256)
k_pack(const float* __restrict__ W1, const float* __restrict__ Wf,
       unsigned short* __restrict__ fragW1, unsigned short* __restrict__ fragWf) {
    int i0 = blockIdx.x * 256 + threadIdx.x;
    const int NW1 = 8 * 8 * 32 * 16;
    for (int idx = i0; idx < NW1; idx += gridDim.x * 256) {
        int e = idx & 15, L = (idx >> 4) & 31, kk = (idx >> 9) & 7, c = idx >> 12;
        int k = kk * 32 + e + 16 * (L >> 4);
        int n = c * 16 + (L & 15);
        fragW1[idx] = f2bf(W1[k * HH + n]);
    }
    const int NWF = 16 * 24 * 32 * 16;
    for (int idx = i0; idx < NWF; idx += gridDim.x * 256) {
        int e = idx & 15, L = (idx >> 4) & 31;
        int kc = idx >> 9;
        int kk = kc % 24, c = kc / 24;
        int k = kk * 32 + e + 16 * (L >> 4);
        int n = c * 16 + (L & 15);
        fragWf[idx] = f2bf(Wf[k * H + n]);
    }
}

// ---- K1: scores = silu(x@W1+b1)@W2 + b2  (WMMA bf16, f32 accum) ------------
// 128 threads = 4 waves; block owns 64 rows; each wave one 16-row tile.
// LDS: 64 rows x 264 bf16 (pad 8 -> row stride 528B, bank stride 4: conflict-free).
__global__ void __launch_bounds__(128)
k_scores(const float* __restrict__ x, const unsigned short* __restrict__ fragW1,
         const float* __restrict__ b1, const float* __restrict__ W2,
         const float* __restrict__ b2, float* __restrict__ scores) {
    __shared__ unsigned short tile[64 * 264];
    const int t = threadIdx.x;
    const int row0 = blockIdx.x * 64;

    // stage 64x256 f32 -> bf16 in LDS (coalesced float4 loads)
    const float4* x4 = (const float4*)(x + (size_t)row0 * H);
    for (int j = t; j < 64 * 64; j += 128) {
        int r = j >> 6, c4 = j & 63;
        float4 v = x4[j];
        unsigned lo = (unsigned)f2bf(v.x) | ((unsigned)f2bf(v.y) << 16);
        unsigned hi = (unsigned)f2bf(v.z) | ((unsigned)f2bf(v.w) << 16);
        *(uint2*)&tile[r * 264 + c4 * 4] = make_uint2(lo, hi);
    }
    __syncthreads();

    const int w = t >> 5, lane = t & 31;
    const int hlf = lane >> 4, n16 = lane & 15;
    const int rloc = w * 16;

    f32x8 acc[8];
#pragma unroll
    for (int c = 0; c < 8; ++c) acc[c] = (f32x8)0.0f;

    const bf16x16* fB = (const bf16x16*)fragW1;
    for (int kk = 0; kk < 8; ++kk) {
        // A fragment (16x32 bf16): lane row m = lane&15; VGPR v in 0..7: K=2v+{0,1}+16*half
        // (contiguous K order, matching the B packing)
        union { unsigned u[8]; bf16x16 vec; } A;
        const unsigned short* arow = &tile[(rloc + n16) * 264 + kk * 32 + hlf * 16];
#pragma unroll
        for (int v = 0; v < 4; ++v) A.u[v] = *(const unsigned*)(arow + 2 * v);
#pragma unroll
        for (int v = 4; v < 8; ++v) A.u[v] = *(const unsigned*)(arow + 8 + 2 * (v - 4));
#pragma unroll
        for (int c = 0; c < 8; ++c) {
            bf16x16 B = fB[(c * 8 + kk) * 32 + lane];
            acc[c] = __builtin_amdgcn_wmma_f32_16x16x32_bf16(
                false, A.vec, false, B, (short)0, acc[c], false, false);
        }
    }

    // hidden -> silu -> dot with W2 (per-row partials, reduce over 16 lanes/half)
    float part[8];
#pragma unroll
    for (int r = 0; r < 8; ++r) part[r] = 0.0f;
#pragma unroll
    for (int c = 0; c < 8; ++c) {
        int col = c * 16 + n16;
        float bb = b1[col], ww = W2[col];
#pragma unroll
        for (int r = 0; r < 8; ++r) {
            float hv = acc[c][r] + bb;
            float sv = hv / (1.0f + expf(-hv));
            part[r] += sv * ww;
        }
    }
#pragma unroll
    for (int r = 0; r < 8; ++r) {
        part[r] += __shfl_xor(part[r], 8, 32);
        part[r] += __shfl_xor(part[r], 4, 32);
        part[r] += __shfl_xor(part[r], 2, 32);
        part[r] += __shfl_xor(part[r], 1, 32);
    }
    if (n16 == 0) {
        float b2v = b2[0];
#pragma unroll
        for (int r = 0; r < 8; ++r)
            scores[row0 + rloc + hlf * 8 + r] = part[r] + b2v;  // C row m = r + 8*half
    }
}

// ---- K2a/K2b: deterministic global softmax stats ---------------------------
__global__ void __launch_bounds__(256)
k_smax_part(const float* __restrict__ scores, int N, float* __restrict__ stats) {
    __shared__ float red[256];
    int t = threadIdx.x;
    int base = blockIdx.x * 1024;
    float m = -3.0e38f;
#pragma unroll
    for (int k = 0; k < 4; ++k) {
        int i = base + t + k * 256;
        if (i < N) m = fmaxf(m, scores[i]);
    }
    red[t] = m; __syncthreads();
    for (int s = 128; s > 0; s >>= 1) { if (t < s) red[t] = fmaxf(red[t], red[t + s]); __syncthreads(); }
    float M = red[0]; __syncthreads();
    float sum = 0.0f;
#pragma unroll
    for (int k = 0; k < 4; ++k) {
        int i = base + t + k * 256;
        if (i < N) sum += expf(scores[i] - M);
    }
    red[t] = sum; __syncthreads();
    for (int s = 128; s > 0; s >>= 1) { if (t < s) red[t] += red[t + s]; __syncthreads(); }
    if (t == 0) { stats[2 * blockIdx.x] = M; stats[2 * blockIdx.x + 1] = red[0]; }
}

__global__ void __launch_bounds__(256)
k_smax_final(const float* __restrict__ stats, float* __restrict__ gstats) {
    __shared__ float red[256];
    int t = threadIdx.x;
    float m0 = stats[2 * t], m1 = stats[2 * (t + 256)];
    red[t] = fmaxf(m0, m1); __syncthreads();
    for (int s = 128; s > 0; s >>= 1) { if (t < s) red[t] = fmaxf(red[t], red[t + s]); __syncthreads(); }
    float M = red[0]; __syncthreads();
    float sum = stats[2 * t + 1] * expf(m0 - M) + stats[2 * (t + 256) + 1] * expf(m1 - M);
    red[t] = sum; __syncthreads();
    for (int s = 128; s > 0; s >>= 1) { if (t < s) red[t] += red[t + s]; __syncthreads(); }
    if (t == 0) { gstats[0] = M; gstats[1] = red[0]; }
}

// ---- K3: per-segment mean/max/attn pooling (batch sorted -> binary search) -
__global__ void __launch_bounds__(256)
k_pool(const float* __restrict__ x, const int* __restrict__ batch,
       const float* __restrict__ scores, const float* __restrict__ gstats,
       int N, float* __restrict__ combined) {
    __shared__ int range[2];
    int g = blockIdx.x, t = threadIdx.x;
    if (t == 0) {
        int lo = 0, hi = N;
        while (lo < hi) { int mid = (lo + hi) >> 1; if (batch[mid] < g) lo = mid + 1; else hi = mid; }
        range[0] = lo;
        hi = N;
        while (lo < hi) { int mid = (lo + hi) >> 1; if (batch[mid] < g + 1) lo = mid + 1; else hi = mid; }
        range[1] = lo;
    }
    __syncthreads();
    int lo = range[0], hi = range[1];
    float M = gstats[0], invS = 1.0f / gstats[1];
    float sum = 0.0f, mx = -3.0e38f, asum = 0.0f;
    for (int i = lo; i < hi; ++i) {
        float v = x[(size_t)i * H + t];
        float wgt = expf(scores[i] - M) * invS;
        sum += v;
        mx = fmaxf(mx, v);
        asum += wgt * v;
    }
    int cnt = hi - lo;
    float mean = sum / (float)(cnt > 0 ? cnt : 1);
    float maxp = cnt > 0 ? mx : 0.0f;
    float* row = combined + (size_t)g * C3H;
    row[t] = mean; row[H + t] = maxp; row[2 * H + t] = asum;
}

// ---- K4: h = combined @ Wf  (WMMA bf16), 256 thr = 8 waves ------------------
// block: 32 rows x 768 K; wave w: row tile w>>2, col group w&3 (64 cols).
__global__ void __launch_bounds__(256)
k_final_gemm(const float* __restrict__ combined, const unsigned short* __restrict__ fragWf,
             float* __restrict__ hout) {
    __shared__ unsigned short tile[32 * 776];   // pad 8 -> 1552B stride, conflict-free
    int t = threadIdx.x;
    int row0 = blockIdx.x * 32;
    const float4* c4p = (const float4*)(combined + (size_t)row0 * C3H);
    for (int j = t; j < 32 * 192; j += 256) {
        int r = j / 192, c4 = j % 192;
        float4 v = c4p[j];
        unsigned lo = (unsigned)f2bf(v.x) | ((unsigned)f2bf(v.y) << 16);
        unsigned hi = (unsigned)f2bf(v.z) | ((unsigned)f2bf(v.w) << 16);
        *(uint2*)&tile[r * 776 + c4 * 4] = make_uint2(lo, hi);
    }
    __syncthreads();

    int w = t >> 5, lane = t & 31, hlf = lane >> 4, n16 = lane & 15;
    int rloc = (w >> 2) * 16, cg = w & 3;
    f32x8 acc[4];
#pragma unroll
    for (int c = 0; c < 4; ++c) acc[c] = (f32x8)0.0f;

    const bf16x16* fB = (const bf16x16*)fragWf;
    for (int kk = 0; kk < 24; ++kk) {
        union { unsigned u[8]; bf16x16 vec; } A;
        const unsigned short* arow = &tile[(rloc + n16) * 776 + kk * 32 + hlf * 16];
#pragma unroll
        for (int v = 0; v < 4; ++v) A.u[v] = *(const unsigned*)(arow + 2 * v);
#pragma unroll
        for (int v = 4; v < 8; ++v) A.u[v] = *(const unsigned*)(arow + 8 + 2 * (v - 4));
#pragma unroll
        for (int c = 0; c < 4; ++c) {
            int ct = cg * 4 + c;
            bf16x16 B = fB[(ct * 24 + kk) * 32 + lane];
            acc[c] = __builtin_amdgcn_wmma_f32_16x16x32_bf16(
                false, A.vec, false, B, (short)0, acc[c], false, false);
        }
    }
#pragma unroll
    for (int c = 0; c < 4; ++c) {
        int col = (cg * 4 + c) * 16 + n16;
#pragma unroll
        for (int r = 0; r < 8; ++r) {
            int row = row0 + rloc + hlf * 8 + r;
            hout[(size_t)row * H + col] = acc[c][r];
        }
    }
}

// ---- K5: LayerNorm + SiLU ---------------------------------------------------
__global__ void __launch_bounds__(256)
k_ln_silu(const float* __restrict__ hin, const float* __restrict__ bfv,
          const float* __restrict__ gamma, const float* __restrict__ beta,
          float* __restrict__ out) {
    __shared__ float red[256];
    __shared__ float red2[256];
    int g = blockIdx.x, t = threadIdx.x;
    float v = hin[(size_t)g * H + t] + bfv[t];
    red[t] = v; red2[t] = v * v; __syncthreads();
    for (int s = 128; s > 0; s >>= 1) {
        if (t < s) { red[t] += red[t + s]; red2[t] += red2[t + s]; }
        __syncthreads();
    }
    float mean = red[0] * (1.0f / H);
    float var  = red2[0] * (1.0f / H) - mean * mean;
    float nv = (v - mean) * rsqrtf(var + 1e-5f) * gamma[t] + beta[t];
    out[(size_t)g * H + t] = nv / (1.0f + expf(-nv));
}

// ---- launch -----------------------------------------------------------------
extern "C" void kernel_launch(void* const* d_in, const int* in_sizes, int n_in,
                              void* d_out, int out_size, void* d_ws, size_t ws_size,
                              hipStream_t stream) {
    const float* x     = (const float*)d_in[0];
    const int*   batch = (const int*)d_in[1];
    const float* W1    = (const float*)d_in[3];
    const float* b1    = (const float*)d_in[4];
    const float* W2    = (const float*)d_in[5];
    const float* b2    = (const float*)d_in[6];
    const float* Wf    = (const float*)d_in[7];
    const float* bfv   = (const float*)d_in[8];
    const float* gamma = (const float*)d_in[9];
    const float* beta  = (const float*)d_in[10];
    float* out = (float*)d_out;

    const int N = in_sizes[0] / H;   // 524288
    const int G = out_size / H;      // 4096

    char* ws = (char*)d_ws;
    size_t off = 0;
    auto alloc = [&](size_t bytes) {
        size_t p = off;
        off = (off + bytes + 255) & ~(size_t)255;
        return p;
    };
    unsigned short* fragW1 = (unsigned short*)(ws + alloc((size_t)8 * 8 * 32 * 16 * 2));
    unsigned short* fragWf = (unsigned short*)(ws + alloc((size_t)16 * 24 * 32 * 16 * 2));
    float* scores   = (float*)(ws + alloc((size_t)N * 4));
    float* stats    = (float*)(ws + alloc((size_t)512 * 2 * 4));
    float* gstats   = (float*)(ws + alloc(2 * 4));
    float* combined = (float*)(ws + alloc((size_t)G * C3H * 4));
    float* hbuf     = (float*)(ws + alloc((size_t)G * H * 4));
    (void)ws_size; (void)n_in;

    k_pack<<<64, 256, 0, stream>>>(W1, Wf, fragW1, fragWf);
    k_scores<<<N / 64, 128, 0, stream>>>(x, fragW1, b1, W2, b2, scores);
    k_smax_part<<<512, 256, 0, stream>>>(scores, N, stats);
    k_smax_final<<<1, 256, 0, stream>>>(stats, gstats);
    k_pool<<<G, 256, 0, stream>>>(x, batch, scores, gstats, N, combined);
    k_final_gemm<<<G / 32, 256, 0, stream>>>(combined, fragWf, hbuf);
    k_ln_silu<<<G, 256, 0, stream>>>(hbuf, bfv, gamma, beta, out);
}
